// SelfAttentionLayer_65481071401231
// MI455X (gfx1250) — matrix operand order repaired and checked by reading, not compiled, not run on hardware
//
#include <hip/hip_runtime.h>
#include <hip/hip_bf16.h>

#define DD 1024
#define BB 256

typedef __attribute__((ext_vector_type(16))) __bf16 v16bf;
typedef __attribute__((ext_vector_type(8)))  float  v8f;

union Frag { v16bf v; uint4 u4[2]; };
struct Frags { Frag a, bq, bk, bv; };

// log2(e) / sqrt(1024)
#define SCALE_C 0.04508422766f

#if defined(__HIP_DEVICE_COMPILE__) && defined(__gfx1250__) && \
    __has_builtin(__builtin_amdgcn_global_load_async_to_lds_b128) && \
    __has_builtin(__builtin_amdgcn_s_wait_asynccnt)
#define HAS_ASYNC_LDS 1
// vector-of-4-int pointer types the async builtin expects
typedef int v4i __attribute__((vector_size(16)));
typedef __attribute__((address_space(1))) v4i g_v4i;
typedef __attribute__((address_space(3))) v4i l_v4i;
#else
#define HAS_ASYNC_LDS 0
#endif

// ---------------- f32 -> bf16 pre-convert ----------------
__global__ __launch_bounds__(256) void cvt_bf16_kernel(const float* __restrict__ src,
                                                       __hip_bfloat16* __restrict__ dst,
                                                       int n) {
    int i = blockIdx.x * blockDim.x + threadIdx.x;
    int stride = gridDim.x * blockDim.x;
    for (; i < n; i += stride) dst[i] = __float2bfloat16(src[i]);
}

// ---------------- fused q/k/v GEMM via WMMA bf16 ----------------
// y = x @ W^T : A = x[256,1024] bf16; W row-major [out][in] => W rows are the
// contraction-contiguous B columns. One wave per 16x16 tile, all 3 matrices,
// register double-buffered K pipeline. k,v stored interleaved as float2.
__device__ __forceinline__ void load_step(Frags& f,
                                          const __hip_bfloat16* arow,
                                          const __hip_bfloat16* qrow,
                                          const __hip_bfloat16* krow,
                                          const __hip_bfloat16* vrow,
                                          int kk, int g) {
    // A 16x32 frag: lane's K chunk [kk+16g, +16)
    f.a.u4[0]  = *(const uint4*)(arow + kk + 16 * g);
    f.a.u4[1]  = *(const uint4*)(arow + kk + 16 * g + 8);
    // B 32x16 frag: lane's K chunk [kk+16g, +16)
    f.bq.u4[0] = *(const uint4*)(qrow + kk + 16 * g);
    f.bq.u4[1] = *(const uint4*)(qrow + kk + 16 * g + 8);
    f.bk.u4[0] = *(const uint4*)(krow + kk + 16 * g);
    f.bk.u4[1] = *(const uint4*)(krow + kk + 16 * g + 8);
    f.bv.u4[0] = *(const uint4*)(vrow + kk + 16 * g);
    f.bv.u4[1] = *(const uint4*)(vrow + kk + 16 * g + 8);
}

__global__ __launch_bounds__(256) void qkv_gemm_kernel(
    const __hip_bfloat16* __restrict__ xbf,
    const __hip_bfloat16* __restrict__ wq,
    const __hip_bfloat16* __restrict__ wk,
    const __hip_bfloat16* __restrict__ wv,
    float* __restrict__ q, float2* __restrict__ kv) {
    const int lane  = threadIdx.x & 31;
    const int wave  = threadIdx.x >> 5;
    const int ntile = blockIdx.x * 8 + wave;   // 0..63
    const int mtile = blockIdx.y;              // 0..15
    const int g     = lane >> 4;               // lane group
    const int mr    = lane & 15;

    const __hip_bfloat16* arow = xbf + (size_t)(mtile * 16 + mr) * DD;
    const __hip_bfloat16* qrow = wq  + (size_t)(ntile * 16 + mr) * DD;
    const __hip_bfloat16* krow = wk  + (size_t)(ntile * 16 + mr) * DD;
    const __hip_bfloat16* vrow = wv  + (size_t)(ntile * 16 + mr) * DD;

    v8f cq = {}; v8f ck = {}; v8f cv = {};

    Frags f[2];
    load_step(f[0], arow, qrow, krow, vrow, 0, g);
#pragma unroll
    for (int s = 0; s < 32; ++s) {
        if (s + 1 < 32)  // prefetch next K-slice into the other register set
            load_step(f[(s + 1) & 1], arow, qrow, krow, vrow, (s + 1) * 32, g);
        Frags& c = f[s & 1];
        cq = __builtin_amdgcn_wmma_f32_16x16x32_bf16(false, c.a.v, false, c.bq.v, (short)0, cq, false, false);
        ck = __builtin_amdgcn_wmma_f32_16x16x32_bf16(false, c.a.v, false, c.bk.v, (short)0, ck, false, false);
        cv = __builtin_amdgcn_wmma_f32_16x16x32_bf16(false, c.a.v, false, c.bv.v, (short)0, cv, false, false);
    }

    // C/D layout: VGPR r -> m = 8*g + r, n = lane&15
    float*  qo  = q  + (size_t)(mtile * 16 + 8 * g) * DD + ntile * 16 + mr;
    float2* kvo = kv + (size_t)(mtile * 16 + 8 * g) * DD + ntile * 16 + mr;
#pragma unroll
    for (int r = 0; r < 8; ++r) {
        qo[r * DD]  = cq[r];
        kvo[r * DD] = make_float2(ck[r], cv[r]);
    }
}

// ---------------- shared attention inner loop ----------------
__device__ __forceinline__ void attn_core(const float2* kvlds, const float a[4],
                                          float kmx, float kmn,
                                          float* __restrict__ out, int b, int tid) {
    float m[4], s[4], acc[4];
#pragma unroll
    for (int ii = 0; ii < 4; ++ii) {
        // max_j a*k_j == max(a*kmax, a*kmin) (sign of a); exact softmax shift
        m[ii]   = fmaxf(a[ii] * kmx, a[ii] * kmn);
        s[ii]   = 0.f;
        acc[ii] = 0.f;
    }
    for (int j = 0; j < DD; ++j) {
        float2 p = kvlds[j];  // ds broadcast: all lanes read same address
#pragma unroll
        for (int ii = 0; ii < 4; ++ii) {
            float t = __builtin_amdgcn_exp2f(__builtin_fmaf(a[ii], p.x, -m[ii]));
            s[ii]   += t;
            acc[ii]  = __builtin_fmaf(t, p.y, acc[ii]);
        }
    }
#pragma unroll
    for (int ii = 0; ii < 4; ++ii)
        out[(size_t)b * DD + tid + 256 * ii] = acc[ii] / s[ii];
}

__device__ __forceinline__ void reduce_minmax(float& kmx, float& kmn, float2* red, int tid) {
#pragma unroll
    for (int off = 16; off; off >>= 1) {
        kmx = fmaxf(kmx, __shfl_xor(kmx, off, 32));
        kmn = fminf(kmn, __shfl_xor(kmn, off, 32));
    }
    if ((tid & 31) == 0) red[tid >> 5] = make_float2(kmx, kmn);
    __syncthreads();
#pragma unroll
    for (int w = 0; w < 8; ++w) {
        kmx = fmaxf(kmx, red[w].x);
        kmn = fminf(kmn, red[w].y);
    }
}

// ---------------- attention from precomputed q, interleaved kv ----------------
__global__ __launch_bounds__(256) void attn_kernel(const float* __restrict__ q,
                                                   const float2* __restrict__ kv,
                                                   float* __restrict__ out) {
    __shared__ float2 kvlds[DD];
    __shared__ float2 red[8];
    const int b = blockIdx.x, tid = threadIdx.x;
    const float2* kvrow = kv + (size_t)b * DD;

    float kmx = -__builtin_inff(), kmn = __builtin_inff();
#if HAS_ASYNC_LDS
    {
        // 8 KB contiguous row copy straight to LDS, ASYNCcnt-tracked.
        // 512 x 16B chunks, 256 lanes x 2.
        g_v4i* gp = (g_v4i*)kvrow;   // builtin takes non-const AS(1) v4i*
        l_v4i* lp = (l_v4i*)kvlds;   // AS(3) v4i*
        __builtin_amdgcn_global_load_async_to_lds_b128(gp + tid, lp + tid, 0, 0);
        __builtin_amdgcn_global_load_async_to_lds_b128(gp + 256 + tid, lp + 256 + tid, 0, 0);
        __builtin_amdgcn_s_wait_asynccnt(0);
    }
    __syncthreads();
#pragma unroll
    for (int t = 0; t < 4; ++t) {
        float kj = kvlds[tid + 256 * t].x;
        kmx = fmaxf(kmx, kj);
        kmn = fminf(kmn, kj);
    }
#else
#pragma unroll
    for (int t = 0; t < 4; ++t) {
        int j = tid + 256 * t;
        float2 p = kvrow[j];
        kvlds[j] = p;
        kmx = fmaxf(kmx, p.x);
        kmn = fminf(kmn, p.x);
    }
#endif
    reduce_minmax(kmx, kmn, red, tid);  // internal __syncthreads covers kvlds visibility

    float a[4];
#pragma unroll
    for (int ii = 0; ii < 4; ++ii)
        a[ii] = q[(size_t)b * DD + tid + 256 * ii] * SCALE_C;

    attn_core(kvlds, a, kmx, kmn, out, b, tid);
}

// ---------------- fully fused fallback (small workspace) ----------------
__global__ __launch_bounds__(256) void attn_fused_kernel(const float* __restrict__ x,
                                                         const float* __restrict__ Wq,
                                                         const float* __restrict__ Wk,
                                                         const float* __restrict__ Wv,
                                                         float* __restrict__ out) {
    __shared__ float  xs[DD];
    __shared__ float2 kvlds[DD];
    __shared__ float2 red[8];
    const int b = blockIdx.x, tid = threadIdx.x;
#pragma unroll
    for (int t = 0; t < 4; ++t) xs[tid + 256 * t] = x[(size_t)b * DD + tid + 256 * t];
    __syncthreads();

    float a[4];
    float kmx = -__builtin_inff(), kmn = __builtin_inff();
#pragma unroll
    for (int ii = 0; ii < 4; ++ii) {
        int i = tid + 256 * ii;
        float dq = 0.f, dk = 0.f, dv = 0.f;
        for (int e = 0; e < DD; ++e) {
            float xe = xs[e];
            dq = __builtin_fmaf(xe, Wq[(size_t)i * DD + e], dq);
            dk = __builtin_fmaf(xe, Wk[(size_t)i * DD + e], dk);
            dv = __builtin_fmaf(xe, Wv[(size_t)i * DD + e], dv);
        }
        a[ii] = dq * SCALE_C;
        kvlds[i] = make_float2(dk, dv);
        kmx = fmaxf(kmx, dk);
        kmn = fminf(kmn, dk);
    }
    __syncthreads();
    reduce_minmax(kmx, kmn, red, tid);
    attn_core(kvlds, a, kmx, kmn, out, b, tid);
}

extern "C" void kernel_launch(void* const* d_in, const int* in_sizes, int n_in,
                              void* d_out, int out_size, void* d_ws, size_t ws_size,
                              hipStream_t stream) {
    const float* x  = (const float*)d_in[0];
    const float* Wq = (const float*)d_in[1];
    const float* Wk = (const float*)d_in[2];
    const float* Wv = (const float*)d_in[3];
    float* out = (float*)d_out;

    const size_t nBD = (size_t)BB * DD;   // 256*1024
    const size_t nDD = (size_t)DD * DD;   // 1024*1024
    const size_t need = (nBD + 3 * nDD) * sizeof(__hip_bfloat16) + 3 * nBD * sizeof(float);

    if (ws_size >= need) {
        char* p = (char*)d_ws;
        __hip_bfloat16* xbf = (__hip_bfloat16*)p;
        __hip_bfloat16* wqb = (__hip_bfloat16*)(p + nBD * 2);
        __hip_bfloat16* wkb = wqb + nDD;
        __hip_bfloat16* wvb = wkb + nDD;
        float*  q  = (float*)(p + (nBD + 3 * nDD) * 2);
        float2* kv = (float2*)(q + nBD);

        cvt_bf16_kernel<<<512, 256, 0, stream>>>(x, xbf, (int)nBD);
        cvt_bf16_kernel<<<2048, 256, 0, stream>>>(Wq, wqb, (int)nDD);
        cvt_bf16_kernel<<<2048, 256, 0, stream>>>(Wk, wkb, (int)nDD);
        cvt_bf16_kernel<<<2048, 256, 0, stream>>>(Wv, wvb, (int)nDD);

        qkv_gemm_kernel<<<dim3(8, 16), 256, 0, stream>>>(xbf, wqb, wkb, wvb, q, kv);
        attn_kernel<<<BB, 256, 0, stream>>>(q, kv, out);
    } else {
        attn_fused_kernel<<<BB, 256, 0, stream>>>(x, Wq, Wk, Wv, out);
    }
}
